// LogSparseLocalMultiHeadAttention_70549132804837
// MI455X (gfx1250) — matrix operand order repaired and checked by reading, not compiled, not run on hardware
//
#include <hip/hip_runtime.h>
#include <math.h>

typedef __attribute__((ext_vector_type(16))) __bf16 v16bf;
typedef __attribute__((ext_vector_type(8)))  float  v8f;

#define NB 2
#define SQ 2048
#define DD 1024
#define NH 16
#define DEP 64
#define MROWS (NB * SQ)          // 4096
#define LOCAL_CTX 32

// native f32 -> bf16 (RNE); pairs fuse into v_cvt_pk_bf16_f32
__device__ __forceinline__ __bf16 f2bf(float x) { return (__bf16)x; }

struct alignas(32) FragBF {
    union {
        v16bf v;
        uint4 q[2];
        __bf16 s[16];
    };
};

// A fragment (16x32 bf16), source row-major with K contiguous.
// lane = (m | khalf<<4); elements 0..7 = K[khalf*8 .. +7], 8..15 = K[16+khalf*8 .. +7]
__device__ __forceinline__ v16bf ld_fragA_bf16(const __bf16* p) {
    FragBF f;
    f.q[0] = *(const uint4*)(p);        // 8 bf16
    f.q[1] = *(const uint4*)(p + 16);   // 8 bf16, 32B further
    return f.v;
}

// B fragment (32x16 bf16), source with K contiguous per column.
// lane = (n | kb<<4); elements 0..7 = K[kb*8 .. +7], 8..15 = K[16+kb*8 .. +7] of column n
__device__ __forceinline__ v16bf ld_fragB_bf16(const __bf16* p) {
    FragBF f;
    f.q[0] = *(const uint4*)(p);
    f.q[1] = *(const uint4*)(p + 16);
    return f.v;
}

__device__ __forceinline__ v8f wmma_bf16(v16bf a, v16bf b, v8f c) {
    return __builtin_amdgcn_wmma_f32_16x16x32_bf16(
        false, a, false, b, (short)0, c, false, false);
}

__device__ __forceinline__ bool allowed_qk(int qi, int ki) {
    int diff = qi - ki;
    if (diff < 0) return false;            // causal
    if (diff <= LOCAL_CTX) return true;    // local window (includes diag)
    float ld = __log2f((float)diff);       // native v_log_f32
    return fabsf(ld - rintf(ld)) < 0.1f;   // log-sparse powers of two (0.1 tol)
}

// ---------------------------------------------------------------------------
// Kernel 1: transpose + convert one 1024x1024 weight matrix to bf16 [N][K]
// ---------------------------------------------------------------------------
__global__ void k_cvt_w(const float* __restrict__ W, __bf16* __restrict__ Wt) {
    int idx = blockIdx.x * 256 + threadIdx.x;   // 0 .. 1M-1
    int n = idx & (DD - 1);
    int k = idx >> 10;
    Wt[(size_t)n * DD + k] = f2bf(W[(size_t)k * DD + n]);
}

// ---------------------------------------------------------------------------
// Kernel 2: Q/K/V projection GEMM.  C[m,n] = X[m,:]@W[:,n] + b[n]
// block: 128 thr (4 waves), tile 16(M) x 64(N); grid.z selects q/k/v.
// Q,K stored bf16 [B,H,S,64]; V stored bf16 transposed [B,H,64,S].
// ---------------------------------------------------------------------------
__global__ void k_proj(const float* __restrict__ Xq, const float* __restrict__ Xk,
                       const float* __restrict__ Xv,
                       const __bf16* __restrict__ Wqt,
                       const __bf16* __restrict__ Wkt,
                       const __bf16* __restrict__ Wvt,
                       const float* __restrict__ bq, const float* __restrict__ bk,
                       const float* __restrict__ bv,
                       __bf16* __restrict__ qh, __bf16* __restrict__ kh,
                       __bf16* __restrict__ vht) {
    const int z = blockIdx.z;
    const float* X = (z == 0) ? Xq : (z == 1) ? Xk : Xv;
    const __bf16* Wt = (z == 0) ? Wqt : (z == 1) ? Wkt : Wvt;
    const float* bias = (z == 0) ? bq : (z == 1) ? bk : bv;

    const int wave = threadIdx.x >> 5;
    const int lane = threadIdx.x & 31;
    const int m0 = blockIdx.x * 16;
    const int n0 = blockIdx.y * 64 + wave * 16;

    const int mrow  = lane & 15;
    const int khalf = (lane >> 4) & 1;
    const int col   = lane & 15;
    const int kb    = (lane >> 4) * 8;
    const int rbase = (lane >> 4) * 8;

    const float* arow = X + (size_t)(m0 + mrow) * DD;
    const __bf16* brow = Wt + (size_t)(n0 + col) * DD;

    v8f acc = {0.f, 0.f, 0.f, 0.f, 0.f, 0.f, 0.f, 0.f};
    for (int kc = 0; kc < DD / 32; ++kc) {
        // A: convert f32 -> bf16 on the fly (two contiguous 8-float chunks)
        FragBF a;
        const float* p0 = arow + kc * 32 + khalf * 8;
        #pragma unroll
        for (int i = 0; i < 8; ++i) a.s[i] = f2bf(p0[i]);
        #pragma unroll
        for (int i = 0; i < 8; ++i) a.s[8 + i] = f2bf(p0[16 + i]);
        // B: contiguous bf16 from transposed weights
        v16bf b = ld_fragB_bf16(brow + kc * 32 + kb);
        acc = wmma_bf16(a.v, b, acc);
    }

    const float bn = bias[n0 + col];
    const int nc = n0 + col;
    const int hh = nc >> 6;       // head
    const int dd = nc & 63;       // depth
    #pragma unroll
    for (int j = 0; j < 8; ++j) {
        const int mm = m0 + rbase + j;
        const int bb = mm >> 11;          // batch (S=2048)
        const int ss = mm & (SQ - 1);
        const __bf16 val = f2bf(acc[j] + bn);
        const size_t headBase = (size_t)(bb * NH + hh);
        if (z == 0)       qh[(headBase * SQ + ss) * DEP + dd] = val;
        else if (z == 1)  kh[(headBase * SQ + ss) * DEP + dd] = val;
        else              vht[(headBase * DEP + dd) * SQ + ss] = val;
    }
}

// ---------------------------------------------------------------------------
// Kernel 3: attention for one (b, h, 16-row q tile).
// 128 thr (4 waves). Dynamic LDS: 16 x 2048 f32 logits (128 KB; WGP has 320 KB).
// Phase 1: QK^T via WMMA, scale + log-sparse mask -> LDS
// Phase 2/3: rowmax / exp+rowsum reductions in LDS
// Phase 4: single write of normalized attn to HBM + P@V via WMMA from LDS
// ---------------------------------------------------------------------------
__global__ void k_attn(const __bf16* __restrict__ qh,
                       const __bf16* __restrict__ kh,
                       const __bf16* __restrict__ vht,
                       float* __restrict__ attnp,
                       __bf16* __restrict__ ctx) {
    extern __shared__ float smem[];                // [16][2048]
    __shared__ float red[128];
    __shared__ float rowmax[16];
    __shared__ float rowinv[16];

    const int tid  = threadIdx.x;
    const int wave = tid >> 5;
    const int lane = tid & 31;
    const int q0 = blockIdx.x * 16;
    const int h  = blockIdx.y;
    const int b  = blockIdx.z;

    const int mrow  = lane & 15;
    const int khalf = (lane >> 4) & 1;
    const int col   = lane & 15;
    const int kb    = (lane >> 4) * 8;
    const int rbase = (lane >> 4) * 8;

    const size_t headBase = (size_t)(b * NH + h);
    const __bf16* qhead = qh + headBase * SQ * DEP;
    const __bf16* khead = kh + headBase * SQ * DEP;
    const __bf16* vhead = vht + headBase * DEP * SQ;

    // ---- Phase 1: logits -> LDS --------------------------------------------
    const __bf16* qrow = qhead + (size_t)(q0 + mrow) * DEP;
    const v16bf a0 = ld_fragA_bf16(qrow + 0  + khalf * 8);   // d 0..31
    const v16bf a1 = ld_fragA_bf16(qrow + 32 + khalf * 8);   // d 32..63

    for (int kt = wave; kt < SQ / 16; kt += 4) {
        const __bf16* krow = khead + (size_t)(kt * 16 + col) * DEP;
        v16bf b0 = ld_fragB_bf16(krow + 0  + kb);
        v16bf b1 = ld_fragB_bf16(krow + 32 + kb);
        v8f acc = {0.f, 0.f, 0.f, 0.f, 0.f, 0.f, 0.f, 0.f};
        acc = wmma_bf16(a0, b0, acc);
        acc = wmma_bf16(a1, b1, acc);
        const int ki = kt * 16 + col;
        #pragma unroll
        for (int j = 0; j < 8; ++j) {
            const int r = rbase + j;
            float vlog = acc[j] * 0.125f;              // / sqrt(64)
            if (!allowed_qk(q0 + r, ki)) vlog += -1e9f;
            smem[r * SQ + ki] = vlog;
        }
    }
    __syncthreads();

    // ---- Phase 2: row max ---------------------------------------------------
    const int r = tid >> 3;
    const int c = tid & 7;
    {
        const float* row = smem + r * SQ + c * 256;
        float m = -1e30f;
        for (int i = 0; i < 256; ++i) m = fmaxf(m, row[i]);
        red[r * 8 + c] = m;
    }
    __syncthreads();
    if (tid < 16) {
        float m = red[tid * 8];
        #pragma unroll
        for (int i = 1; i < 8; ++i) m = fmaxf(m, red[tid * 8 + i]);
        rowmax[tid] = m;
    }
    __syncthreads();

    // ---- Phase 3: exp + row sum --------------------------------------------
    {
        float* row = smem + r * SQ + c * 256;
        const float mx = rowmax[r];
        float s = 0.f;
        for (int i = 0; i < 256; ++i) {
            float e = expf(row[i] - mx);
            row[i] = e;
            s += e;
        }
        red[r * 8 + c] = s;
    }
    __syncthreads();
    if (tid < 16) {
        float s = red[tid * 8];
        #pragma unroll
        for (int i = 1; i < 8; ++i) s += red[tid * 8 + i];
        rowinv[tid] = 1.0f / s;
    }
    __syncthreads();

    // ---- Phase 4a: normalize in LDS + single streaming write of attn -------
    {
        float* row = smem + r * SQ + c * 256;
        float4* dst = (float4*)(attnp + ((headBase * SQ + (q0 + r)) * (size_t)SQ) + c * 256);
        const float inv = rowinv[r];
        float4* row4 = (float4*)row;
        for (int i = 0; i < 64; ++i) {
            float4 pv = row4[i];
            pv.x *= inv; pv.y *= inv; pv.z *= inv; pv.w *= inv;
            row4[i] = pv;
            dst[i] = pv;
        }
    }
    __syncthreads();

    // ---- Phase 4b: ctx = P @ V ; wave w owns d-tile [w*16 .. +15] ----------
    {
        const int n0 = wave * 16;
        v8f acc = {0.f, 0.f, 0.f, 0.f, 0.f, 0.f, 0.f, 0.f};
        const float* prow = smem + mrow * SQ;             // probs row for A frag
        const __bf16* vcol = vhead + (size_t)(n0 + col) * SQ;
        for (int kc = 0; kc < SQ / 32; ++kc) {
            FragBF a;
            const float* p0 = prow + kc * 32 + khalf * 8;
            #pragma unroll
            for (int i = 0; i < 8; ++i) a.s[i] = f2bf(p0[i]);
            #pragma unroll
            for (int i = 0; i < 8; ++i) a.s[8 + i] = f2bf(p0[16 + i]);
            v16bf bf = ld_fragB_bf16(vcol + kc * 32 + kb);
            acc = wmma_bf16(a.v, bf, acc);
        }
        #pragma unroll
        for (int j = 0; j < 8; ++j) {
            const int qq = q0 + rbase + j;
            ctx[(size_t)(b * SQ + qq) * DD + h * DEP + n0 + col] = f2bf(acc[j]);
        }
    }
}

// ---------------------------------------------------------------------------
// Kernel 4: output projection  out = ctx @ Wo + bo   (f32 out)
// ---------------------------------------------------------------------------
__global__ void k_outproj(const __bf16* __restrict__ ctx,
                          const __bf16* __restrict__ Wot,
                          const float* __restrict__ bo,
                          float* __restrict__ outp) {
    const int wave = threadIdx.x >> 5;
    const int lane = threadIdx.x & 31;
    const int m0 = blockIdx.x * 16;
    const int n0 = blockIdx.y * 64 + wave * 16;

    const int mrow  = lane & 15;
    const int khalf = (lane >> 4) & 1;
    const int col   = lane & 15;
    const int kb    = (lane >> 4) * 8;
    const int rbase = (lane >> 4) * 8;

    const __bf16* arow = ctx + (size_t)(m0 + mrow) * DD;
    const __bf16* brow = Wot + (size_t)(n0 + col) * DD;

    v8f acc = {0.f, 0.f, 0.f, 0.f, 0.f, 0.f, 0.f, 0.f};
    for (int kc = 0; kc < DD / 32; ++kc) {
        v16bf a = ld_fragA_bf16(arow + kc * 32 + khalf * 8);
        v16bf b = ld_fragB_bf16(brow + kc * 32 + kb);
        acc = wmma_bf16(a, b, acc);
    }
    const float bn = bo[n0 + col];
    #pragma unroll
    for (int j = 0; j < 8; ++j) {
        outp[(size_t)(m0 + rbase + j) * DD + n0 + col] = acc[j] + bn;
    }
}

// ---------------------------------------------------------------------------
extern "C" void kernel_launch(void* const* d_in, const int* in_sizes, int n_in,
                              void* d_out, int out_size, void* d_ws, size_t ws_size,
                              hipStream_t stream) {
    const float* v_in = (const float*)d_in[0];
    const float* k_in = (const float*)d_in[1];
    const float* q_in = (const float*)d_in[2];
    const float* Wq = (const float*)d_in[3];
    const float* bq = (const float*)d_in[4];
    const float* Wk = (const float*)d_in[5];
    const float* bk = (const float*)d_in[6];
    const float* Wv = (const float*)d_in[7];
    const float* bv = (const float*)d_in[8];
    const float* Wo = (const float*)d_in[9];
    const float* bo = (const float*)d_in[10];

    float* outp  = (float*)d_out;
    float* attnp = outp + (size_t)NB * SQ * DD;

    // workspace carve (bytes)
    char* w = (char*)d_ws;
    const size_t szHead = (size_t)NB * SQ * DD * sizeof(unsigned short);  // 8 MB
    const size_t szW    = (size_t)DD * DD * sizeof(unsigned short);       // 2 MB
    __bf16* qh  = (__bf16*)(w);
    __bf16* kh  = (__bf16*)(w + szHead);
    __bf16* vht = (__bf16*)(w + 2 * szHead);
    __bf16* Wqt = (__bf16*)(w + 3 * szHead);
    __bf16* Wkt = (__bf16*)(w + 3 * szHead + szW);
    __bf16* Wvt = (__bf16*)(w + 3 * szHead + 2 * szW);
    __bf16* Wot = (__bf16*)(w + 3 * szHead + 3 * szW);
    __bf16* ctx = (__bf16*)(w + 3 * szHead + 4 * szW);

    // 1) weight transpose + bf16 convert
    k_cvt_w<<<(DD * DD) / 256, 256, 0, stream>>>(Wq, Wqt);
    k_cvt_w<<<(DD * DD) / 256, 256, 0, stream>>>(Wk, Wkt);
    k_cvt_w<<<(DD * DD) / 256, 256, 0, stream>>>(Wv, Wvt);
    k_cvt_w<<<(DD * DD) / 256, 256, 0, stream>>>(Wo, Wot);

    // 2) Q/K/V projections (z picks which)
    k_proj<<<dim3(MROWS / 16, DD / 64, 3), 128, 0, stream>>>(
        q_in, k_in, v_in, Wqt, Wkt, Wvt, bq, bk, bv, qh, kh, vht);

    // 3) attention: (qtile, head, batch); 128 KB dynamic LDS
    k_attn<<<dim3(SQ / 16, NH, NB), 128, 16 * SQ * sizeof(float), stream>>>(
        qh, kh, vht, attnp, ctx);

    // 4) output projection
    k_outproj<<<dim3(MROWS / 16, DD / 64, 1), 128, 0, stream>>>(ctx, Wot, bo, outp);
}
